// MegatronWindowAttention_17927193494180
// MI455X (gfx1250) — matrix operand order repaired and checked by reading, not compiled, not run on hardware
//
#include <hip/hip_runtime.h>

// ---------------------------------------------------------------------------
// Swin window attention, MI455X (gfx1250, wave32, WMMA bf16 16x16x32 + TDM).
//   4096 windows x 49 tokens x 256 dim, 8 heads x 32 head_dim.
// Pipeline:
//   [1] bias gather  -> ws.bias  [8][49][49] f32
//   [2] QKV GEMM     -> ws.qkv   [200704][768] bf16
//       x tile staged HBM->LDS by the Tensor Data Mover (tensor_load_to_lds,
//       TENSORcnt), fragments converted f32->bf16 at LDS read time.
//   [3] fused attention + projection per window -> d_out f32
// ---------------------------------------------------------------------------

#define NWIN   4096
#define NTOK   49
#define DIMC   256
#define NHEAD  8
#define HDIM   32
#define MTOT   (NWIN * NTOK)          // 200704 rows, divisible by 32
#define QKVN   768
#define SCALE  0.17677669529663687f   // 1/sqrt(32)

typedef __attribute__((ext_vector_type(16))) __bf16 v16bf;
typedef __attribute__((ext_vector_type(8)))  __bf16 v8bf;
typedef __attribute__((ext_vector_type(8)))  float  v8f;
typedef __attribute__((ext_vector_type(4)))  unsigned int uint4v;
typedef __attribute__((ext_vector_type(8)))  unsigned int uint8v;

union AB16 { v16bf v; __bf16 e[16]; };

static __device__ __forceinline__ v8f zero8f() {
    v8f z;
#pragma unroll
    for (int i = 0; i < 8; ++i) z[i] = 0.f;
    return z;
}

static __device__ __forceinline__ v16bf zero16bf() {
    AB16 u;
#pragma unroll
    for (int i = 0; i < 16; ++i) u.e[i] = (__bf16)0.f;
    return u.v;
}

// 16 contiguous bf16 (B-fragment chunk: K = e + 16*half handled by caller base)
static __device__ __forceinline__ v16bf ld16(const __bf16* p) {
    AB16 u;
    *(v8bf*)(&u.e[0]) = *(const v8bf*)(p);
    *(v8bf*)(&u.e[8]) = *(const v8bf*)(p + 8);
    return u.v;
}

// two contiguous 8-element chunks (A-fragment: K = e + 16*half)
static __device__ __forceinline__ v16bf ld8x2(const __bf16* p0, const __bf16* p1) {
    AB16 u;
    *(v8bf*)(&u.e[0]) = *(const v8bf*)(p0);
    *(v8bf*)(&u.e[8]) = *(const v8bf*)(p1);
    return u.v;
}

// 16 contiguous f32 -> bf16 B-fragment chunk
static __device__ __forceinline__ v16bf cvt16(const float* p) {
    AB16 u;
#pragma unroll
    for (int i = 0; i < 16; ++i) u.e[i] = (__bf16)p[i];
    return u.v;
}

// two 8-float chunks (f32 LDS) -> bf16 A-fragment
static __device__ __forceinline__ v16bf cvt8x2(const float* p0, const float* p1) {
    AB16 u;
#pragma unroll
    for (int i = 0; i < 8; ++i) u.e[i]     = (__bf16)p0[i];
#pragma unroll
    for (int i = 0; i < 8; ++i) u.e[8 + i] = (__bf16)p1[i];
    return u.v;
}

static __device__ __forceinline__ v8f wmma_bf16(v16bf a, v16bf b, v8f c) {
    // D = A(16x32) * B(32x16) + C(16x16 f32)
    return __builtin_amdgcn_wmma_f32_16x16x32_bf16(
        /*neg_a=*/false, a, /*neg_b=*/false, b,
        /*c_mod=*/(short)0, c, /*reuse_a=*/false, /*reuse_b=*/false);
}

// ---------------------------------------------------------------------------
// [1] relative-position bias gather: biasw[h][q][n] = table[rel_idx(q,n)*8 + h]
// ---------------------------------------------------------------------------
__global__ __launch_bounds__(256) void bias_gather_kernel(
        const float* __restrict__ table, float* __restrict__ biasw) {
    int i = blockIdx.x * 256 + threadIdx.x;
    if (i >= NHEAD * NTOK * NTOK) return;
    int h   = i / (NTOK * NTOK);
    int rem = i % (NTOK * NTOK);
    int q = rem / NTOK, n = rem % NTOK;
    int dh = (q / 7) - (n / 7) + 6;          // + (WH-1)
    int dw = (q % 7) - (n % 7) + 6;          // + (WW-1)
    biasw[i] = table[(dh * 13 + dw) * NHEAD + h];   // *(2*WW-1)
}

// ---------------------------------------------------------------------------
// [2] QKV GEMM: qkv[m][n] = sum_k x[m][k] * qkv_w[n][k] + qkv_b[n], bf16 out.
// Workgroup: 32 M-rows x full N=768 (x streamed from HBM exactly once).
// The 32x32 f32 x-tile is DMA'd into LDS by the Tensor Data Mover each
// K-step; wave 0 issues the descriptor and waits on TENSORcnt.
// 8 waves, wave w owns N cols [w*96, w*96+96); per wave 2 Mf x 6 Nf tiles.
// ---------------------------------------------------------------------------
__global__ __launch_bounds__(256) void qkv_gemm_kernel(
        const float* __restrict__ x, const float* __restrict__ w,
        const float* __restrict__ bias, __bf16* __restrict__ qkv) {
    __shared__ float sA[32][32];             // 32 rows x 32 K, f32 (TDM target)

    const int tid  = threadIdx.x;
    const int wv   = tid >> 5;
    const int lane = tid & 31;
    const int half = lane >> 4;
    const int l16  = lane & 15;
    const int mbase = blockIdx.x * 32;

    // Uniform pieces of the tensor DMA descriptor (D#), ISA 8.3/8.4.
    const unsigned int lds_addr = (unsigned int)(uintptr_t)&sA[0][0]; // LDS byte offset
    const unsigned long long gbase =
        (unsigned long long)(uintptr_t)x + (unsigned long long)mbase * DIMC * 4ull;

    v8f acc[2][6];
#pragma unroll
    for (int mf = 0; mf < 2; ++mf)
#pragma unroll
        for (int nf = 0; nf < 6; ++nf) acc[mf][nf] = zero8f();

    for (int kb = 0; kb < DIMC; kb += 32) {
        __syncthreads();                     // previous tile fully consumed
        if (wv == 0) {
            unsigned long long ga = gbase + (unsigned long long)kb * 4ull;
            // D# group 0: count=1 | lds_addr | global_addr[56:0] | type=2
            uint4v g0;
            g0[0] = 1u;                                       // count=1, user mode
            g0[1] = lds_addr;
            g0[2] = (unsigned int)(ga & 0xFFFFFFFFull);
            g0[3] = (unsigned int)((ga >> 32) & 0x1FFFFFFull) | (2u << 30); // type=2
            // D# group 1: data_size=4B, tensor 256 x MTOT, tile 32x32,
            //             tensor_dim0_stride=256 (elements)
            uint8v g1;
            g1[0] = (2u << 16);                               // data_size=2 -> 4B
            g1[1] = (256u & 0xFFFFu) << 16;                   // tensor_dim0[15:0]
            g1[2] = ((256u >> 16) & 0xFFFFu)                  // tensor_dim0[31:16]
                  | (((unsigned)MTOT & 0xFFFFu) << 16);       // tensor_dim1[15:0]
            g1[3] = (((unsigned)MTOT >> 16) & 0xFFFFu)        // tensor_dim1[31:16]
                  | (32u << 16);                              // tile_dim0=32
            g1[4] = 32u;                                      // tile_dim1=32, tile_dim2=0
            g1[5] = 256u;                                     // dim0_stride[31:0]
            g1[6] = 0u;                                       // dim0_stride[47:32], dim1_stride lo
            g1[7] = 0u;                                       // dim1_stride hi
            asm volatile("tensor_load_to_lds %0, %1"
                         :: "s"(g0), "s"(g1)
                         : "memory");
            __builtin_amdgcn_s_wait_tensorcnt(0);
        }
        __syncthreads();                     // tile visible to all waves

        v16bf afr[2];
#pragma unroll
        for (int mf = 0; mf < 2; ++mf) {
            int row = mf * 16 + l16;
            afr[mf] = cvt8x2(&sA[row][16 * half], &sA[row][16 * half + 8]);
        }
#pragma unroll
        for (int nf = 0; nf < 6; ++nf) {
            int n = wv * 96 + nf * 16 + l16;
            v16bf bfr = cvt16(w + (size_t)n * DIMC + kb + 16 * half);
            acc[0][nf] = wmma_bf16(afr[0], bfr, acc[0][nf]);
            acc[1][nf] = wmma_bf16(afr[1], bfr, acc[1][nf]);
        }
    }

#pragma unroll
    for (int mf = 0; mf < 2; ++mf)
#pragma unroll
        for (int nf = 0; nf < 6; ++nf) {
            int n = wv * 96 + nf * 16 + l16;
            float bn = bias[n];
#pragma unroll
            for (int r = 0; r < 8; ++r) {
                int m = mbase + mf * 16 + r + 8 * half;
                qkv[(size_t)m * QKVN + n] = (__bf16)(acc[mf][nf][r] + bn);
            }
        }
}

// ---------------------------------------------------------------------------
// [3] fused attention + projection. One block per window, one wave per head.
//   S = q k^T * scale + bias + mask (pad 49->64), softmax, O = P v -> sO(LDS)
//   then the whole block does out = sO @ proj_w^T + proj_b.
// ---------------------------------------------------------------------------
__global__ __launch_bounds__(256) void attn_proj_kernel(
        const __bf16* __restrict__ qkv, const float* __restrict__ biasw,
        const float* __restrict__ mask, const float* __restrict__ pw,
        const float* __restrict__ pb, float* __restrict__ out) {
    __shared__ __bf16 sP[NHEAD][16 * 64];    // per-wave P strip (16 rows x 64)
    __shared__ __bf16 sO[64][DIMC];          // window attention output (padded)

    const int b    = blockIdx.x;
    const int tid  = threadIdx.x;
    const int h    = tid >> 5;               // wave == head
    const int lane = tid & 31;
    const int half = lane >> 4;
    const int l16  = lane & 15;

    const __bf16* qp = qkv + (size_t)b * NTOK * QKVN + h * HDIM;
    const __bf16* kp = qp + DIMC;            // k at col offset 256
    const __bf16* vp = qp + 2 * DIMC;        // v at col offset 512

    // K as B-fragments: lane -> key token, elements -> contiguous head dims.
    v16bf kf[4];
#pragma unroll
    for (int nt = 0; nt < 4; ++nt) {
        int n = nt * 16 + l16;
        kf[nt] = (n < NTOK) ? ld16(kp + (size_t)n * QKVN + 16 * half) : zero16bf();
    }
    // V as B-fragments: lane -> head dim, elements -> key tokens (strided).
    v16bf vf[2][2];
#pragma unroll
    for (int kt = 0; kt < 2; ++kt)
#pragma unroll
        for (int dt = 0; dt < 2; ++dt) {
            AB16 u;
            int d = dt * 16 + l16;
#pragma unroll
            for (int e = 0; e < 16; ++e) {
                int t = kt * 32 + e + 16 * half;
                u.e[e] = (t < NTOK) ? vp[(size_t)t * QKVN + d] : (__bf16)0.f;
            }
            vf[kt][dt] = u.v;
        }

    const float* bw = biasw + h * NTOK * NTOK;
    const float* mk = mask + (size_t)(b & 63) * NTOK * NTOK;   // b % nW

    for (int mt = 0; mt < 4; ++mt) {
        // Q A-fragment for this 16-row block
        int m0 = mt * 16 + l16;
        v16bf qf = (m0 < NTOK)
            ? ld8x2(qp + (size_t)m0 * QKVN + 16 * half,
                    qp + (size_t)m0 * QKVN + 16 * half + 8)
            : zero16bf();

        v8f s[4];
#pragma unroll
        for (int nt = 0; nt < 4; ++nt) {
            s[nt] = zero8f();
            s[nt] = wmma_bf16(qf, kf[nt], s[nt]);   // K=32 covers head_dim
        }
        // scale + rel-pos bias + shifted-window mask + key padding
#pragma unroll
        for (int nt = 0; nt < 4; ++nt) {
            int n  = nt * 16 + l16;
            int ni = (n < NTOK) ? n : NTOK - 1;
#pragma unroll
            for (int r = 0; r < 8; ++r) {
                int m  = mt * 16 + r + 8 * half;
                int mi = (m < NTOK) ? m : NTOK - 1;
                float v = s[nt][r] * SCALE + bw[mi * NTOK + ni] + mk[mi * NTOK + ni];
                if (n >= NTOK) v = -1e30f;
                s[nt][r] = v;
            }
        }
        // softmax per row (row lives in one 16-lane group, 4 regs across tiles)
#pragma unroll
        for (int r = 0; r < 8; ++r) {
            float mx = s[0][r];
            mx = fmaxf(mx, s[1][r]); mx = fmaxf(mx, s[2][r]); mx = fmaxf(mx, s[3][r]);
#pragma unroll
            for (int off = 8; off >= 1; off >>= 1)
                mx = fmaxf(mx, __shfl_xor(mx, off, 32));
            float sum = 0.f;
            float ex[4];
#pragma unroll
            for (int nt = 0; nt < 4; ++nt) { ex[nt] = __expf(s[nt][r] - mx); sum += ex[nt]; }
#pragma unroll
            for (int off = 8; off >= 1; off >>= 1)
                sum += __shfl_xor(sum, off, 32);
            float inv = 1.f / sum;
            int rowl = r + 8 * half;
#pragma unroll
            for (int nt = 0; nt < 4; ++nt)
                sP[h][rowl * 64 + nt * 16 + l16] = (__bf16)(ex[nt] * inv);
        }
        // O = P @ V   (per-wave private LDS strip; DS ops in-order per wave)
        v8f oa[2] = { zero8f(), zero8f() };
#pragma unroll
        for (int kt = 0; kt < 2; ++kt) {
            const __bf16* pr = &sP[h][l16 * 64 + kt * 32];
            v16bf pf = ld8x2(pr + 16 * half, pr + 16 * half + 8);
            oa[0] = wmma_bf16(pf, vf[kt][0], oa[0]);
            oa[1] = wmma_bf16(pf, vf[kt][1], oa[1]);
        }
#pragma unroll
        for (int dt = 0; dt < 2; ++dt)
#pragma unroll
            for (int r = 0; r < 8; ++r)
                sO[mt * 16 + r + 8 * half][h * HDIM + dt * 16 + l16] = (__bf16)oa[dt][r];
    }

    __syncthreads();   // all heads finished writing sO

    // projection: out = sO(64x256) @ proj_w^T(256x256) + proj_b
    // wave h owns output cols [h*32, h*32+32); 4 Mf x 2 Nf tiles.
    v8f pacc[4][2];
#pragma unroll
    for (int mf = 0; mf < 4; ++mf) { pacc[mf][0] = zero8f(); pacc[mf][1] = zero8f(); }

    for (int kb = 0; kb < DIMC; kb += 32) {
        v16bf afr[4];
#pragma unroll
        for (int mf = 0; mf < 4; ++mf) {
            int row = mf * 16 + l16;
            afr[mf] = ld8x2(&sO[row][kb + 16 * half], &sO[row][kb + 16 * half + 8]);
        }
#pragma unroll
        for (int nf = 0; nf < 2; ++nf) {
            int n = h * HDIM + nf * 16 + l16;
            v16bf bfr = cvt16(pw + (size_t)n * DIMC + kb + 16 * half);
#pragma unroll
            for (int mf = 0; mf < 4; ++mf)
                pacc[mf][nf] = wmma_bf16(afr[mf], bfr, pacc[mf][nf]);
        }
    }
#pragma unroll
    for (int mf = 0; mf < 4; ++mf)
#pragma unroll
        for (int nf = 0; nf < 2; ++nf) {
            int n = h * HDIM + nf * 16 + l16;
            float bn = pb[n];
#pragma unroll
            for (int r = 0; r < 8; ++r) {
                int m = mf * 16 + r + 8 * half;
                if (m < NTOK)
                    out[((size_t)b * NTOK + m) * DIMC + n] = pacc[mf][nf][r] + bn;
            }
        }
}

// ---------------------------------------------------------------------------
extern "C" void kernel_launch(void* const* d_in, const int* in_sizes, int n_in,
                              void* d_out, int out_size, void* d_ws, size_t ws_size,
                              hipStream_t stream) {
    (void)in_sizes; (void)n_in; (void)out_size; (void)ws_size;
    const float* x          = (const float*)d_in[0];
    const float* mask       = (const float*)d_in[1];
    const float* qkv_w      = (const float*)d_in[2];
    const float* qkv_b      = (const float*)d_in[3];
    const float* proj_w     = (const float*)d_in[4];
    const float* proj_b     = (const float*)d_in[5];
    const float* bias_table = (const float*)d_in[6];

    // workspace layout: [qkv bf16: 200704*768*2 B][bias f32: 8*49*49*4 B]
    __bf16* qkvws  = (__bf16*)d_ws;
    float*  biasws = (float*)((char*)d_ws + (size_t)MTOT * QKVN * sizeof(__bf16));

    bias_gather_kernel<<<(NHEAD * NTOK * NTOK + 255) / 256, 256, 0, stream>>>(
        bias_table, biasws);
    qkv_gemm_kernel<<<MTOT / 32, 256, 0, stream>>>(x, qkv_w, qkv_b, qkvws);
    attn_proj_kernel<<<NWIN, 256, 0, stream>>>(qkvws, biasws, mask,
                                               proj_w, proj_b, (float*)d_out);
}
